// CausalSelfAttention_53360673686170
// MI455X (gfx1250) — matrix operand order repaired and checked by reading, not compiled, not run on hardware
//
#include <hip/hip_runtime.h>
#include <hip/hip_bf16.h>

typedef __bf16 bf16;
typedef __attribute__((ext_vector_type(16))) __bf16 v16bf;
typedef __attribute__((ext_vector_type(8)))  __bf16 v8bf;
typedef __attribute__((ext_vector_type(4)))  __bf16 v4bf;
typedef __attribute__((ext_vector_type(8)))  float  v8f;

#define E_DIM 1024
#define S_LEN 2048
#define NH    16
#define HD    64

// A-fragment (16-bit, 16x32): lane-half h reads K runs [8h..8h+7] and [16+8h..23+8h]
// B-fragment (16-bit, 32x16): lane-half h reads K runs [8h..8h+7] and [16+8h..23+8h]
// C/D        (f32, 16x16):    reg j <-> row j+8h, lane&15 <-> col

__device__ __forceinline__ v16bf frag2(const bf16* p0, const bf16* p1) {
    v8bf lo = *(const v8bf*)p0;
    v8bf hi = *(const v8bf*)p1;
    return __builtin_shufflevector(lo, hi, 0,1,2,3,4,5,6,7,8,9,10,11,12,13,14,15);
}

__device__ __forceinline__ v8f wmma_bf16(v16bf a, v16bf b, v8f c) {
    return __builtin_amdgcn_wmma_f32_16x16x32_bf16(false, a, false, b, (short)0, c, false, false);
}

__device__ __forceinline__ void async_ld_b128(const bf16* gptr, bf16* lptr) {
    unsigned long long ga = (unsigned long long)(const void*)gptr;
    unsigned la = (unsigned)(unsigned long long)(void*)lptr;   // low 32 bits = LDS offset
    asm volatile("global_load_async_to_lds_b128 %0, %1, off"
                 : : "v"(la), "v"(ga) : "memory");
}
__device__ __forceinline__ void wait_async0() {
    asm volatile("s_wait_asynccnt 0x0" ::: "memory");
}

// ---------------------------------------------------------------------------
// Kernel 1: QKV projection GEMM  Y = X * W  -> bf16 [B][H][S][D] (Q scaled)
// grid = (M/128, N/128, 3), block = 256 (8 waves; 4x2 wave grid, 32x64/wave)
// ---------------------------------------------------------------------------
__global__ __launch_bounds__(256)
void qkv_gemm(const float* __restrict__ X,
              const float* __restrict__ Wq, const float* __restrict__ Wk,
              const float* __restrict__ Wv,
              bf16* __restrict__ Qb, bf16* __restrict__ Kb, bf16* __restrict__ Vb)
{
    const int which = blockIdx.z;
    const float* W = (which == 0) ? Wq : ((which == 1) ? Wk : Wv);
    bf16* dst      = (which == 0) ? Qb : ((which == 1) ? Kb : Vb);
    const float scale = (which == 0) ? 0.125f : 1.0f;   // 1/sqrt(64) folded into Q

    __shared__ __align__(16) bf16 As [128][40];   // [m][k]   80B stride
    __shared__ __align__(16) bf16 BsT[128][40];   // [n][k]   transposed weight tile

    const int tid  = threadIdx.x;
    const int lane = tid & 31;
    const int wave = tid >> 5;
    const int half = (lane >> 4) & 1;
    const int l16  = lane & 15;

    const int m0 = blockIdx.x * 128;
    const int n0 = blockIdx.y * 128;
    const int wm = (wave & 3) * 32;
    const int wn = (wave >> 2) * 64;

    v8f acc[2][4];
#pragma unroll
    for (int mt = 0; mt < 2; ++mt)
#pragma unroll
        for (int nt = 0; nt < 4; ++nt)
#pragma unroll
            for (int j = 0; j < 8; ++j) acc[mt][nt][j] = 0.0f;

    for (int k0 = 0; k0 < E_DIM; k0 += 32) {
        // A tile 128x32 (f32 -> bf16, packed b64 stores)
#pragma unroll
        for (int p = 0; p < 4; ++p) {
            int r = p * 32 + (tid >> 3);
            int c = (tid & 7) * 4;
            const float4 f = *(const float4*)(X + (size_t)(m0 + r) * E_DIM + k0 + c);
            v4bf t; t[0] = (bf16)f.x; t[1] = (bf16)f.y; t[2] = (bf16)f.z; t[3] = (bf16)f.w;
            *(v4bf*)&As[r][c] = t;
        }
        // B tile 32x128 (f32 -> bf16) stored transposed [n][k]
#pragma unroll
        for (int p = 0; p < 4; ++p) {
            int r = p * 8 + (tid >> 5);          // k
            int c = (tid & 31) * 4;              // n
            const float4 f = *(const float4*)(W + (size_t)(k0 + r) * E_DIM + n0 + c);
            BsT[c + 0][r] = (bf16)f.x; BsT[c + 1][r] = (bf16)f.y;
            BsT[c + 2][r] = (bf16)f.z; BsT[c + 3][r] = (bf16)f.w;
        }
        __syncthreads();

        v16bf af[2];
#pragma unroll
        for (int mt = 0; mt < 2; ++mt) {
            int m = wm + mt * 16 + l16;
            af[mt] = frag2(&As[m][8 * half], &As[m][16 + 8 * half]);
        }
#pragma unroll
        for (int nt = 0; nt < 4; ++nt) {
            int n = wn + nt * 16 + l16;
            v16bf bfrag = frag2(&BsT[n][8 * half], &BsT[n][16 + 8 * half]);
            acc[0][nt] = wmma_bf16(af[0], bfrag, acc[0][nt]);
            acc[1][nt] = wmma_bf16(af[1], bfrag, acc[1][nt]);
        }
        __syncthreads();
    }

    // epilog: bf16 into [B][H][S][D]
#pragma unroll
    for (int mt = 0; mt < 2; ++mt)
#pragma unroll
        for (int nt = 0; nt < 4; ++nt)
#pragma unroll
            for (int j = 0; j < 8; ++j) {
                int gm = m0 + wm + mt * 16 + j + 8 * half;   // token index
                int gn = n0 + wn + nt * 16 + l16;            // embed index
                int b = gm >> 11, s = gm & (S_LEN - 1);
                int h = gn >> 6,  d = gn & (HD - 1);
                dst[(((size_t)b * NH + h) * S_LEN + s) * HD + d] =
                    (bf16)(acc[mt][nt][j] * scale);
            }
}

// ---------------------------------------------------------------------------
// Kernel 2: flash attention. grid = (B*H, S/128), block = 256.
// Wave w handles q rows [qb*128 + w*16, +16); streams K/V in 32-key blocks.
// K tile: async global->LDS DMA (row-major matches B-frag reads).
// V tile: manual load + transposed store (VsT[d][key]) for P*V B-frags.
// ---------------------------------------------------------------------------
__global__ __launch_bounds__(256)
void attn_kernel(const bf16* __restrict__ Qb, const bf16* __restrict__ Kb,
                 const bf16* __restrict__ Vb, bf16* __restrict__ Cxt)
{
    const int bh = blockIdx.x;                  // 0..31
    const int qb = blockIdx.y;                  // 0..15
    const int b = bh >> 4, h = bh & 15;
    const bf16* Qh = Qb + (size_t)bh * S_LEN * HD;
    const bf16* Kh = Kb + (size_t)bh * S_LEN * HD;
    const bf16* Vh = Vb + (size_t)bh * S_LEN * HD;

    __shared__ __align__(16) bf16 Ks [32][72];      // [key][d]  144B stride
    __shared__ __align__(16) bf16 VsT[64][40];      // [d][key]  transposed
    __shared__ __align__(16) bf16 Ps [8][16][40];   // per-wave P transpose scratch

    const int tid  = threadIdx.x;
    const int lane = tid & 31;
    const int wave = tid >> 5;
    const int half = (lane >> 4) & 1;
    const int l16  = lane & 15;
    const int q0   = qb * 128 + wave * 16;

    // Q fragments (two A-frags covering d=0..63), vectorized global loads
    v16bf qf[2];
#pragma unroll
    for (int part = 0; part < 2; ++part) {
        const bf16* qrow = Qh + (size_t)(q0 + l16) * HD + part * 32;
        qf[part] = frag2(qrow + 8 * half, qrow + 16 + 8 * half);
    }

    v8f cacc[4];
#pragma unroll
    for (int nt = 0; nt < 4; ++nt)
#pragma unroll
        for (int j = 0; j < 8; ++j) cacc[nt][j] = 0.0f;
    float rmax[8], rsum[8];
#pragma unroll
    for (int j = 0; j < 8; ++j) { rmax[j] = -INFINITY; rsum[j] = 0.0f; }

    const int nkb = qb * 4 + 4;                 // key blocks of 32 up to causal edge
    for (int kb = 0; kb < nkb; ++kb) {
        const int kstart = kb * 32;
        {
            int r = tid >> 3;                   // key row 0..31
            int c = (tid & 7) * 8;              // d offset, 8 bf16 = 16B
            // K: async DMA straight into LDS (row-major is what K-frags read)
            async_ld_b128(Kh + (size_t)(kstart + r) * HD + c, &Ks[r][c]);
            // V: load + transpose into VsT[d][key]
            const v8bf vv = *(const v8bf*)(Vh + (size_t)(kstart + r) * HD + c);
#pragma unroll
            for (int u = 0; u < 8; ++u) VsT[c + u][r] = vv[u];
        }
        wait_async0();
        __syncthreads();

        if (kstart <= q0 + 15) {                // at least one valid key for this wave
            // scores: two 16x16 tiles (keys kstart+l16, kstart+16+l16)
            v8f s0v, s1v;
#pragma unroll
            for (int j = 0; j < 8; ++j) { s0v[j] = 0.0f; s1v[j] = 0.0f; }
#pragma unroll
            for (int part = 0; part < 2; ++part) {
                const bf16* kr0 = &Ks[l16][part * 32 + 8 * half];
                const bf16* kr1 = &Ks[16 + l16][part * 32 + 8 * half];
                v16bf kf0 = frag2(kr0, kr0 + 16);
                v16bf kf1 = frag2(kr1, kr1 + 16);
                s0v = wmma_bf16(qf[part], kf0, s0v);
                s1v = wmma_bf16(qf[part], kf1, s1v);
            }

            // causal mask + online softmax update (per row j -> q = q0+j+8*half)
            float corr[8];
#pragma unroll
            for (int j = 0; j < 8; ++j) {
                int q   = q0 + j + 8 * half;
                int kg0 = kstart + l16;
                float a = (kg0      <= q) ? s0v[j] : -INFINITY;
                float c = (kg0 + 16 <= q) ? s1v[j] : -INFINITY;
                float mx = fmaxf(a, c);
#pragma unroll
                for (int off = 1; off < 16; off <<= 1)
                    mx = fmaxf(mx, __shfl_xor(mx, off, 32));
                float nm = fmaxf(rmax[j], mx);
                float e0 = __expf(a - nm);
                float e1 = __expf(c - nm);
                float cr = __expf(rmax[j] - nm);
                rmax[j] = nm;
                float ps = e0 + e1;
#pragma unroll
                for (int off = 1; off < 16; off <<= 1)
                    ps += __shfl_xor(ps, off, 32);
                rsum[j] = rsum[j] * cr + ps;
                corr[j] = cr;
                // stash probabilities through LDS (transpose into A-frag layout)
                Ps[wave][j + 8 * half][l16]      = (bf16)e0;
                Ps[wave][j + 8 * half][16 + l16] = (bf16)e1;
            }
            // rescale running context
#pragma unroll
            for (int nt = 0; nt < 4; ++nt)
#pragma unroll
                for (int j = 0; j < 8; ++j) cacc[nt][j] *= corr[j];

            // P as A-fragment (wave-private LDS; same-wave DS ops are in order)
            v16bf pf = frag2(&Ps[wave][l16][8 * half], &Ps[wave][l16][16 + 8 * half]);

            // ctx += P * V  (4 n-tiles over d)
#pragma unroll
            for (int nt = 0; nt < 4; ++nt) {
                int d = nt * 16 + l16;
                v16bf vf = frag2(&VsT[d][8 * half], &VsT[d][16 + 8 * half]);
                cacc[nt] = wmma_bf16(pf, vf, cacc[nt]);
            }
        }
        __syncthreads();
    }

    // finalize: ctx /= rowsum, write bf16 [B][S][E] with heads merged
#pragma unroll
    for (int nt = 0; nt < 4; ++nt)
#pragma unroll
        for (int j = 0; j < 8; ++j) {
            int q = q0 + j + 8 * half;
            int d = nt * 16 + l16;
            float v = cacc[nt][j] / rsum[j];
            Cxt[((size_t)(b * S_LEN + q)) * E_DIM + h * HD + d] = (bf16)v;
        }
}

// ---------------------------------------------------------------------------
// Kernel 3: output projection  out = ctx(bf16) * Wo + bo  -> f32
// ---------------------------------------------------------------------------
__global__ __launch_bounds__(256)
void out_gemm(const bf16* __restrict__ Cx, const float* __restrict__ Wo,
              const float* __restrict__ bo, float* __restrict__ Y)
{
    __shared__ __align__(16) bf16 As [128][40];
    __shared__ __align__(16) bf16 BsT[128][40];

    const int tid  = threadIdx.x;
    const int lane = tid & 31;
    const int wave = tid >> 5;
    const int half = (lane >> 4) & 1;
    const int l16  = lane & 15;

    const int m0 = blockIdx.x * 128;
    const int n0 = blockIdx.y * 128;
    const int wm = (wave & 3) * 32;
    const int wn = (wave >> 2) * 64;

    v8f acc[2][4];
#pragma unroll
    for (int mt = 0; mt < 2; ++mt)
#pragma unroll
        for (int nt = 0; nt < 4; ++nt)
#pragma unroll
            for (int j = 0; j < 8; ++j) acc[mt][nt][j] = 0.0f;

    for (int k0 = 0; k0 < E_DIM; k0 += 32) {
        // A tile 128x32 bf16 (direct 16B copies)
#pragma unroll
        for (int p = 0; p < 2; ++p) {
            int r = p * 64 + (tid >> 2);
            int c = (tid & 3) * 8;
            *(v8bf*)&As[r][c] = *(const v8bf*)(Cx + (size_t)(m0 + r) * E_DIM + k0 + c);
        }
        // B tile 32x128 (f32 -> bf16) stored transposed [n][k]
#pragma unroll
        for (int p = 0; p < 4; ++p) {
            int r = p * 8 + (tid >> 5);
            int c = (tid & 31) * 4;
            const float4 f = *(const float4*)(Wo + (size_t)(k0 + r) * E_DIM + n0 + c);
            BsT[c + 0][r] = (bf16)f.x; BsT[c + 1][r] = (bf16)f.y;
            BsT[c + 2][r] = (bf16)f.z; BsT[c + 3][r] = (bf16)f.w;
        }
        __syncthreads();

        v16bf af[2];
#pragma unroll
        for (int mt = 0; mt < 2; ++mt) {
            int m = wm + mt * 16 + l16;
            af[mt] = frag2(&As[m][8 * half], &As[m][16 + 8 * half]);
        }
#pragma unroll
        for (int nt = 0; nt < 4; ++nt) {
            int n = wn + nt * 16 + l16;
            v16bf bfrag = frag2(&BsT[n][8 * half], &BsT[n][16 + 8 * half]);
            acc[0][nt] = wmma_bf16(af[0], bfrag, acc[0][nt]);
            acc[1][nt] = wmma_bf16(af[1], bfrag, acc[1][nt]);
        }
        __syncthreads();
    }

#pragma unroll
    for (int mt = 0; mt < 2; ++mt)
#pragma unroll
        for (int nt = 0; nt < 4; ++nt)
#pragma unroll
            for (int j = 0; j < 8; ++j) {
                int gm = m0 + wm + mt * 16 + j + 8 * half;
                int gn = n0 + wn + nt * 16 + l16;
                Y[(size_t)gm * E_DIM + gn] = acc[mt][nt][j] + bo[gn];
            }
}

// ---------------------------------------------------------------------------
extern "C" void kernel_launch(void* const* d_in, const int* in_sizes, int n_in,
                              void* d_out, int out_size, void* d_ws, size_t ws_size,
                              hipStream_t stream) {
    const float* x  = (const float*)d_in[0];
    const float* Wq = (const float*)d_in[1];
    const float* Wk = (const float*)d_in[2];
    const float* Wv = (const float*)d_in[3];
    const float* Wo = (const float*)d_in[4];
    const float* bo = (const float*)d_in[5];
    float* out = (float*)d_out;

    const size_t perBuf = (size_t)2 * NH * S_LEN * HD;   // 4,194,304 bf16 elems
    bf16* q  = (bf16*)d_ws;
    bf16* k  = q + perBuf;
    bf16* v  = k + perBuf;
    bf16* cx = v + perBuf;

    dim3 g1(32, 8, 3);                 // M=4096/128, N=1024/128, {Q,K,V}
    qkv_gemm<<<g1, 256, 0, stream>>>(x, Wq, Wk, Wv, q, k, v);

    dim3 g2(2 * NH, S_LEN / 128);      // (b,h) x q-blocks
    attn_kernel<<<g2, 256, 0, stream>>>(q, k, v, cx);

    dim3 g3(32, 8);
    out_gemm<<<g3, 256, 0, stream>>>(cx, Wo, bo, out);
}
